// LSTMDecoder_36790689857735
// MI455X (gfx1250) — matrix operand order repaired and checked by reading, not compiled, not run on hardware
//
#include <hip/hip_runtime.h>

// CDNA5 / gfx1250 WMMA types
typedef __attribute__((ext_vector_type(16))) __bf16 v16bf;
typedef __attribute__((ext_vector_type(8)))  float  v8f;

#define B_DIM 64
#define H_DIM 1024

#define WMMA_BF16(accv, av, bv)                                              \
  accv = __builtin_amdgcn_wmma_f32_16x16x32_bf16(false, av, false, bv,       \
                                                 (short)0, accv, false, false)

union FragU { v16bf v; uint4 q[2]; };

// A fragment chunks: 16 bytes at +0 and 16 bytes at +16B (K+8 elements).
__device__ __forceinline__ v16bf load_a_at(const __bf16* pa) {
  FragU u;
  u.q[0] = *(const uint4*)(pa);
  u.q[1] = *(const uint4*)(pa + 8);
  return u.v;
}
// B fragment: 32 contiguous bytes of one row of row-major W (one W^T column).
__device__ __forceinline__ v16bf load_b_at(const __bf16* pb) {
  FragU u;
  u.q[0] = ((const uint4*)pb)[0];
  u.q[1] = ((const uint4*)pb)[1];
  return u.v;
}

// K=1024 GEMM core, 2-stage ping-pong software pipeline. One 16x64 slab per
// wave (4 WMMA tiles). Stage fragments keep stable registers (no rotation
// copies); all intra-stage deltas are immediate vmem offsets; only two
// loop-carried pointer increments per iteration.
__device__ __forceinline__ void gemm_k1024(const __bf16* A, int arow,
                                           const __bf16* W, int ncol0,
                                           int lane, v8f acc[4]) {
  const int hi = lane >> 4;
  const __bf16* pa = A + arow * 1024 + (hi << 4);   // +16 K for upper half
  const __bf16* pb = W + ncol0 * 1024 + (hi << 4);  // +16 K for upper half

  v16bf a0  = load_a_at(pa);
  v16bf b00 = load_b_at(pb);
  v16bf b01 = load_b_at(pb + 16 * 1024);
  v16bf b02 = load_b_at(pb + 32 * 1024);
  v16bf b03 = load_b_at(pb + 48 * 1024);
  v16bf a1  = load_a_at(pa + 32);
  v16bf b10 = load_b_at(pb + 32);
  v16bf b11 = load_b_at(pb + 16 * 1024 + 32);
  v16bf b12 = load_b_at(pb + 32 * 1024 + 32);
  v16bf b13 = load_b_at(pb + 48 * 1024 + 32);

  for (int it = 0; it < 15; ++it) {   // K-blocks 2..31 (2 per iteration)
    pa += 64; pb += 64;
    WMMA_BF16(acc[0], a0, b00);
    WMMA_BF16(acc[1], a0, b01);
    WMMA_BF16(acc[2], a0, b02);
    WMMA_BF16(acc[3], a0, b03);
    a0  = load_a_at(pa);
    b00 = load_b_at(pb);
    b01 = load_b_at(pb + 16 * 1024);
    b02 = load_b_at(pb + 32 * 1024);
    b03 = load_b_at(pb + 48 * 1024);
    WMMA_BF16(acc[0], a1, b10);
    WMMA_BF16(acc[1], a1, b11);
    WMMA_BF16(acc[2], a1, b12);
    WMMA_BF16(acc[3], a1, b13);
    a1  = load_a_at(pa + 32);
    b10 = load_b_at(pb + 32);
    b11 = load_b_at(pb + 16 * 1024 + 32);
    b12 = load_b_at(pb + 32 * 1024 + 32);
    b13 = load_b_at(pb + 48 * 1024 + 32);
  }
  WMMA_BF16(acc[0], a0, b00);
  WMMA_BF16(acc[1], a0, b01);
  WMMA_BF16(acc[2], a0, b02);
  WMMA_BF16(acc[3], a0, b03);
  WMMA_BF16(acc[0], a1, b10);
  WMMA_BF16(acc[1], a1, b11);
  WMMA_BF16(acc[2], a1, b12);
  WMMA_BF16(acc[3], a1, b13);
}

__device__ __forceinline__ float sigf(float x) { return 1.0f / (1.0f + __expf(-x)); }
// tanh via the v_exp_f32 TRANS path: tanh(x) = 1 - 2/(e^{2x}+1)
__device__ __forceinline__ float tanh_fast(float x) {
  return 1.0f - 2.0f / (__expf(2.0f * x) + 1.0f);
}

// GroupNorm epilogue over one gate group (16 rows x 1024 cols per workgroup).
// Lane-half butterfly first (xor 8..1 stays inside each 16-lane half, which
// owns a disjoint M-row set), then 1 atomic per row per half-wave.
__device__ __forceinline__ void groupnorm_epilogue(
    v8f acc[4], const float bv[4], const float gv[4], const float btv[4],
    float* s_sum, float* s_sq, float* zn, int mrow0, int n0, int nl, int hi) {
#pragma unroll
  for (int t = 0; t < 4; ++t)
#pragma unroll
    for (int j = 0; j < 8; ++j) acc[t][j] += bv[t];

#pragma unroll
  for (int j = 0; j < 8; ++j) {
    float s = 0.0f, ss = 0.0f;
#pragma unroll
    for (int t = 0; t < 4; ++t) { float v = acc[t][j]; s += v; ss += v * v; }
#pragma unroll
    for (int m = 8; m >= 1; m >>= 1) {
      s  += __shfl_xor(s, m, 32);
      ss += __shfl_xor(ss, m, 32);
    }
    if (nl == 0) {
      atomicAdd(&s_sum[hi * 8 + j], s);
      atomicAdd(&s_sq[hi * 8 + j], ss);
    }
  }
  __syncthreads();

  const float inv = 1.0f / 1024.0f;
#pragma unroll
  for (int j = 0; j < 8; ++j) {
    int   ml   = hi * 8 + j;
    float mean = s_sum[ml] * inv;
    float var  = s_sq[ml] * inv - mean * mean;    // biased (torch GroupNorm)
    float rstd = rsqrtf(var + 1e-5f);
    int   mg   = mrow0 + ml;
#pragma unroll
    for (int t = 0; t < 4; ++t) {
      int nc = n0 + t * 16 + nl;
      zn[mg * 4096 + nc] = (acc[t][j] - mean) * rstd * gv[t] + btv[t];
    }
  }
}

// K1: z = A(bf16, global) @ W^T + bias -> GroupNorm -> zn.
// Grid 16 = 4 M-blocks x 4 gates; block 512 = 16 wave32s, 64 cols/wave.
__global__ __launch_bounds__(512) void gemm_groupnorm_kernel(
    const __bf16* __restrict__ A, const __bf16* __restrict__ W,
    const float* __restrict__ bias, const float* __restrict__ gamma,
    const float* __restrict__ beta, float* __restrict__ zn) {
  __shared__ float s_sum[16];
  __shared__ float s_sq[16];
  const int lane = threadIdx.x & 31, wave = threadIdx.x >> 5;
  const int mblk = blockIdx.x & 3, gate = blockIdx.x >> 2;
  const int hi = lane >> 4, nl = lane & 15;
  const int n0 = gate * 1024 + wave * 64;
  const int arow = mblk * 16 + nl;

  float bv[4], gv[4], btv[4];
#pragma unroll
  for (int t = 0; t < 4; ++t) {
    bv[t]  = bias[n0 + t * 16 + nl];
    gv[t]  = gamma[n0 + t * 16 + nl];
    btv[t] = beta[n0 + t * 16 + nl];
  }
  if (threadIdx.x < 16) { s_sum[threadIdx.x] = 0.0f; s_sq[threadIdx.x] = 0.0f; }
  __syncthreads();

  v8f acc[4] = {v8f{}, v8f{}, v8f{}, v8f{}};
  gemm_k1024(A, arow, W, n0 + nl, lane, acc);
  groupnorm_epilogue(acc, bv, gv, btv, s_sum, s_sq, zn, mblk * 16, n0, nl, hi);
}

// K2: fused gate-update + GEMM + GroupNorm. Prologue recomputes
// h = pointwise(zn_in, c_in) for this M-block's 16 rows (gate-duplicated,
// bit-identical), writes c_out, stages h bf16 in LDS; GEMM reads A from LDS.
__global__ __launch_bounds__(512) void lstm_fused_cell_gemm_kernel(
    const float* __restrict__ zn_in, const float* __restrict__ c_in,
    float* __restrict__ c_out, const __bf16* __restrict__ W,
    const float* __restrict__ bias, const float* __restrict__ gamma,
    const float* __restrict__ beta, float* __restrict__ zn_out) {
  __shared__ __bf16 lds_h[16 * 1024];   // 32 KB A-tile
  __shared__ float s_sum[16];
  __shared__ float s_sq[16];
  const int lane = threadIdx.x & 31, wave = threadIdx.x >> 5;
  const int mblk = blockIdx.x & 3, gate = blockIdx.x >> 2;
  const int hi = lane >> 4, nl = lane & 15;
  const int n0 = gate * 1024 + wave * 64;
  const int mrow0 = mblk * 16;

#pragma unroll 4
  for (int it = 0; it < 32; ++it) {
    int i = threadIdx.x + it * 512;
    int r = i >> 10, h = i & 1023;
    int gm = mrow0 + r;
    const float* z = zn_in + gm * 4096;
    float gi = z[h], gf = z[1024 + h], go = z[2048 + h], gg = z[3072 + h];
    float c  = c_in[gm * 1024 + h];
    float cn = sigf(gf) * c + sigf(gi) * tanh_fast(gg);
    float hn = sigf(go) * tanh_fast(cn);
    c_out[gm * 1024 + h] = cn;  // duplicated identical stores across gate-WGs
    lds_h[i] = (__bf16)hn;
  }
  float bv[4], gv[4], btv[4];
#pragma unroll
  for (int t = 0; t < 4; ++t) {
    bv[t]  = bias[n0 + t * 16 + nl];
    gv[t]  = gamma[n0 + t * 16 + nl];
    btv[t] = beta[n0 + t * 16 + nl];
  }
  if (threadIdx.x < 16) { s_sum[threadIdx.x] = 0.0f; s_sq[threadIdx.x] = 0.0f; }
  __syncthreads();

  v8f acc[4] = {v8f{}, v8f{}, v8f{}, v8f{}};
  gemm_k1024(lds_h, nl, W, n0 + nl, lane, acc);   // A fragments from LDS
  groupnorm_epilogue(acc, bv, gv, btv, s_sum, s_sq, zn_out, mrow0, n0, nl, hi);
}

// K3: fused gate-update + output GEMM. Grid 4 (one WG per M-block); writes
// f32 output slice and the bf16 recursive decoder input.
__global__ __launch_bounds__(512) void lstm_fused_out_gemm_kernel(
    const float* __restrict__ zn_in, const float* __restrict__ c_in,
    float* __restrict__ c_out, const __bf16* __restrict__ W,
    const float* __restrict__ bias, float* __restrict__ out_f,
    __bf16* __restrict__ out_b) {
  __shared__ __bf16 lds_h[16 * 1024];
  const int lane = threadIdx.x & 31, wave = threadIdx.x >> 5;
  const int mblk = blockIdx.x;
  const int hi = lane >> 4, nl = lane & 15;
  const int n0 = wave * 64;
  const int mrow0 = mblk * 16;

#pragma unroll 4
  for (int it = 0; it < 32; ++it) {
    int i = threadIdx.x + it * 512;
    int r = i >> 10, h = i & 1023;
    int gm = mrow0 + r;
    const float* z = zn_in + gm * 4096;
    float gi = z[h], gf = z[1024 + h], go = z[2048 + h], gg = z[3072 + h];
    float c  = c_in[gm * 1024 + h];
    float cn = sigf(gf) * c + sigf(gi) * tanh_fast(gg);
    float hn = sigf(go) * tanh_fast(cn);
    c_out[gm * 1024 + h] = cn;
    lds_h[i] = (__bf16)hn;
  }
  float bv[4];
#pragma unroll
  for (int t = 0; t < 4; ++t) bv[t] = bias[n0 + t * 16 + nl];
  __syncthreads();

  v8f acc[4] = {v8f{}, v8f{}, v8f{}, v8f{}};
  gemm_k1024(lds_h, nl, W, n0 + nl, lane, acc);

#pragma unroll
  for (int j = 0; j < 8; ++j) {
    int m = mrow0 + hi * 8 + j;
#pragma unroll
    for (int t = 0; t < 4; ++t) {
      int   nc = n0 + t * 16 + nl;
      float v  = acc[t][j] + bv[t];
      out_f[m * 1024 + nc] = v;
      out_b[m * 1024 + nc] = (__bf16)v;
    }
  }
}

__global__ void f32_to_bf16_kernel(const float* __restrict__ src,
                                   __bf16* __restrict__ dst, int n) {
  int i = blockIdx.x * blockDim.x + threadIdx.x;
  if (i < n) dst[i] = (__bf16)src[i];
}

__global__ void init_state_kernel(const float* __restrict__ h0,
                                  const float* __restrict__ c0,
                                  __bf16* __restrict__ dec_in,
                                  float* __restrict__ c, int n) {
  int i = blockIdx.x * blockDim.x + threadIdx.x;
  if (i < n) { dec_in[i] = (__bf16)h0[i]; c[i] = c0[i]; }
}

extern "C" void kernel_launch(void* const* d_in, const int* in_sizes, int n_in,
                              void* d_out, int out_size, void* d_ws, size_t ws_size,
                              hipStream_t stream) {
  const float* h0   = (const float*)d_in[0];
  const float* c0   = (const float*)d_in[1];
  const float* W1   = (const float*)d_in[2];
  const float* b1   = (const float*)d_in[3];
  const float* g1   = (const float*)d_in[4];
  const float* bt1  = (const float*)d_in[5];
  const float* W2   = (const float*)d_in[6];
  const float* b2   = (const float*)d_in[7];
  const float* g2   = (const float*)d_in[8];
  const float* bt2  = (const float*)d_in[9];
  const float* Wout = (const float*)d_in[10];
  const float* bout = (const float*)d_in[11];
  float* out = (float*)d_out;

  const int T = out_size / (B_DIM * H_DIM);   // 256

  // Workspace layout (~21.6 MB, 256B-aligned offsets).
  char* ws = (char*)d_ws;
  __bf16* W1b   = (__bf16*)(ws);                  // 8 MB
  __bf16* W2b   = (__bf16*)(ws + 8388608);        // 8 MB
  __bf16* Woutb = (__bf16*)(ws + 16777216);       // 2 MB
  float*  zn1   = (float*) (ws + 18874368);       // 1 MB
  float*  zn2   = (float*) (ws + 19922944);       // 1 MB
  float*  cA    = (float*) (ws + 20971520);       // 256 KB (carried c)
  float*  cB    = (float*) (ws + 21233664);       // 256 KB (c after lstm1)
  __bf16* decb  = (__bf16*)(ws + 21495808);       // 128 KB recursive input
  (void)in_sizes; (void)n_in; (void)ws_size;

  // Per-launch weight quantization to bf16 (deterministic; L2-resident after).
  f32_to_bf16_kernel<<<(4194304 + 255) / 256, 256, 0, stream>>>(W1, W1b, 4194304);
  f32_to_bf16_kernel<<<(4194304 + 255) / 256, 256, 0, stream>>>(W2, W2b, 4194304);
  f32_to_bf16_kernel<<<(1048576 + 255) / 256, 256, 0, stream>>>(Wout, Woutb, 1048576);
  init_state_kernel<<<(65536 + 255) / 256, 256, 0, stream>>>(h0, c0, decb, cA, 65536);

  for (int t = 0; t < T; ++t) {
    gemm_groupnorm_kernel<<<16, 512, 0, stream>>>(decb, W1b, b1, g1, bt1, zn1);
    lstm_fused_cell_gemm_kernel<<<16, 512, 0, stream>>>(zn1, cA, cB, W2b,
                                                        b2, g2, bt2, zn2);
    lstm_fused_out_gemm_kernel<<<4, 512, 0, stream>>>(
        zn2, cB, cA, Woutb, bout, out + (size_t)t * (B_DIM * H_DIM), decb);
  }
}
